// RNNModel_59622736003142
// MI455X (gfx1250) — compile-verified, matrix operand order checked by reading, not run on hardware
//
#include <hip/hip_runtime.h>
#include <hip/hip_bf16.h>
#include <math.h>

// ---------------------------------------------------------------------------
// Problem constants (from the reference)
// ---------------------------------------------------------------------------
#define S_LEN   64
#define BATCH   16
#define NTOK    10000
#define NCLASS  4
#define NINP    512
#define NHID    1024
#define NGATE   (4 * NHID)          // 4096
#define ROWS    (S_LEN * BATCH)     // 1024 rows of "out"

typedef __attribute__((ext_vector_type(8)))  _Float16 h8;
typedef __attribute__((ext_vector_type(16))) _Float16 v16h;
typedef __attribute__((ext_vector_type(8)))  float    v8f;

// ---------------------------------------------------------------------------
// WMMA f16 operand loader (global memory).
// 16-bit 16x32 operand layout (CDNA5 ISA 7.12.2):
//   lane l: row = l&15, kbase = (l>>4)*8
//   elements 0..7  -> K = kblock + kbase + 0..7
//   elements 8..15 -> K = kblock + kbase + 16..23
// ---------------------------------------------------------------------------
__device__ __forceinline__ v16h ld_tile(const _Float16* __restrict__ rowbase,
                                        int kblock, int lane) {
  const int kb = ((lane >> 4) << 3);
  const _Float16* p = rowbase + kblock + kb;
  h8 lo = *(const h8*)(p);
  h8 hi = *(const h8*)(p + 16);
  return __builtin_shufflevector(lo, hi, 0, 1, 2, 3, 4, 5, 6, 7,
                                         8, 9, 10, 11, 12, 13, 14, 15);
}

// Same operand pattern, but from a row-major 16x32 tile staged in LDS.
__device__ __forceinline__ v16h ld_tile_lds(const _Float16* tile, int lane) {
  const int r15 = lane & 15;
  const int kb  = ((lane >> 4) << 3);
  const _Float16* p = tile + r15 * 32 + kb;
  h8 lo = *(const h8*)(p);
  h8 hi = *(const h8*)(p + 16);
  return __builtin_shufflevector(lo, hi, 0, 1, 2, 3, 4, 5, 6, 7,
                                         8, 9, 10, 11, 12, 13, 14, 15);
}

__device__ __forceinline__ float sigmoidf_(float x) {
  return 1.0f / (1.0f + expf(-x));
}

// ---------------------------------------------------------------------------
// CDNA5 async global->LDS copy of one 16x32 f16 A-tile (1 KB) issued by one
// wave: 64 16-byte chunks, 32 lanes x 2 instructions.  Tracked by ASYNCcnt.
// ---------------------------------------------------------------------------
__device__ __forceinline__ void async_copy_a_tile(const _Float16* gbase,
                                                  int ld, int kblock,
                                                  _Float16* lbuf, int lane) {
#pragma unroll
  for (int q = 0; q < 2; ++q) {
    int c   = lane + 32 * q;     // chunk 0..63
    int row = c >> 2;            // 0..15
    int sub = c & 3;             // 4 chunks of 8 f16 per row
    const void* g = gbase + (size_t)row * ld + kblock + sub * 8;
    unsigned loff = (unsigned)(uintptr_t)(lbuf + c * 8);  // LDS byte offset
    asm volatile("global_load_async_to_lds_b128 %0, %1, off"
                 :: "v"(loff), "v"(g) : "memory");
  }
}

__device__ __forceinline__ void wait_async_le2() {
  asm volatile("s_wait_asynccnt 0x2" ::: "memory");
}
__device__ __forceinline__ void wait_async_0() {
  asm volatile("s_wait_asynccnt 0x0" ::: "memory");
}

// ---------------------------------------------------------------------------
// Generic f32 -> f16 conversion (weight down-convert, done once per call)
// ---------------------------------------------------------------------------
__global__ void f2h_kernel(const float* __restrict__ src,
                           _Float16* __restrict__ dst, int n) {
  int i = blockIdx.x * blockDim.x + threadIdx.x;
  if (i < n) dst[i] = (_Float16)src[i];
}

__global__ void bias_sum_kernel(const float* __restrict__ a,
                                const float* __restrict__ b,
                                float* __restrict__ o, int n) {
  int i = blockIdx.x * blockDim.x + threadIdx.x;
  if (i < n) o[i] = a[i] + b[i];
}

// ---------------------------------------------------------------------------
// Embedding gather -> f16 activations. x[s,b,k] = emb[words[s,b], k]
// ---------------------------------------------------------------------------
__global__ void embed_kernel(const int* __restrict__ words,
                             const float* __restrict__ emb,
                             _Float16* __restrict__ xh) {
  int i = blockIdx.x * blockDim.x + threadIdx.x;       // S*B*NINP
  if (i >= S_LEN * BATCH * NINP) return;
  int sb = i / NINP;
  int k  = i - sb * NINP;
  int tok = words[sb];
  xh[i] = (_Float16)emb[(size_t)tok * NINP + k];
}

// ---------------------------------------------------------------------------
// Initialize LSTM state from h0 / c0 inputs ((2,B,NHID) each).
// ---------------------------------------------------------------------------
__global__ void init_state_kernel(const float* __restrict__ h0,
                                  const float* __restrict__ c0,
                                  float* __restrict__ Hf,
                                  float* __restrict__ Cf,
                                  _Float16* __restrict__ Hh /*[2][2][B*NHID]*/) {
  int i = blockIdx.x * blockDim.x + threadIdx.x;       // 2*B*NHID = 32768
  if (i >= 2 * BATCH * NHID) return;
  Hf[i] = h0[i];
  Cf[i] = c0[i];
  int layer  = i >> 14;            // / (B*NHID)
  int within = i & 16383;
  Hh[(size_t)layer * 2 * BATCH * NHID + within] = (_Float16)h0[i];
}

// ---------------------------------------------------------------------------
// One LSTM timestep with split-K.
// Block = 128 threads = 4 waves, 64 blocks; block b owns hidden units
// [b*16, b*16+16).  Wave w computes the partial i/f/g/o gate tiles over its
// quarter of the K reduction (WMMA), partials are summed through LDS, and the
// 128 threads apply the fused LSTM cell update.  h(f16) is double-buffered
// across steps (other blocks may still read h_{t-1}).
// ---------------------------------------------------------------------------
__global__ void lstm_step_kernel(const _Float16* __restrict__ xt,  int Kx,
                                 const _Float16* __restrict__ Wih, // (NGATE,Kx) f16
                                 const _Float16* __restrict__ Whh, // (NGATE,NHID) f16
                                 const _Float16* __restrict__ hprev, // (B,NHID) f16
                                 const float*    __restrict__ bias,  // (NGATE)
                                 float*          __restrict__ C,     // (B,NHID)
                                 float*          __restrict__ Hf,    // (B,NHID)
                                 _Float16*       __restrict__ Hnext, // (B,NHID) f16
                                 _Float16*       __restrict__ ys_h,  // step slice f16
                                 float*          __restrict__ ys_f)  // step slice f32 (nullable)
{
  __shared__ float part[4][4][256];   // [wave][gate][m*16+n] partial gates

  const int lane = threadIdx.x & 31;
  const int wv   = threadIdx.x >> 5;        // 0..3 : K-split index
  const int n0   = blockIdx.x * 16;         // hidden-unit tile base
  const int r15  = lane & 15;

  v8f acc[4];
#pragma unroll
  for (int g = 0; g < 4; ++g) acc[g] = (v8f){0, 0, 0, 0, 0, 0, 0, 0};

  // gates += x_t @ Wih^T   (this wave's quarter of Kx)
  {
    const int kchunk = Kx >> 2;                 // 128 or 256
    const int k0 = wv * kchunk, k1 = k0 + kchunk;
    const _Float16* arow = xt + (size_t)r15 * Kx;
    for (int kb = k0; kb < k1; kb += 32) {
      v16h a = ld_tile(arow, kb, lane);
#pragma unroll
      for (int g = 0; g < 4; ++g) {
        const _Float16* brow = Wih + (size_t)(g * NHID + n0 + r15) * Kx;
        v16h b = ld_tile(brow, kb, lane);
        acc[g] = __builtin_amdgcn_wmma_f32_16x16x32_f16(
            false, a, false, b, (short)0, acc[g], false, false);
      }
    }
  }

  // gates += h_{t-1} @ Whh^T   (this wave's quarter of NHID)
  {
    const int k0 = wv * (NHID >> 2), k1 = k0 + (NHID >> 2);
    const _Float16* hrow = hprev + (size_t)r15 * NHID;
    for (int kb = k0; kb < k1; kb += 32) {
      v16h a = ld_tile(hrow, kb, lane);
#pragma unroll
      for (int g = 0; g < 4; ++g) {
        const _Float16* brow = Whh + (size_t)(g * NHID + n0 + r15) * NHID;
        v16h b = ld_tile(brow, kb, lane);
        acc[g] = __builtin_amdgcn_wmma_f32_16x16x32_f16(
            false, a, false, b, (short)0, acc[g], false, false);
      }
    }
  }

  // Scatter partial accumulators to LDS. Accumulator element r in lane l is
  // (m = r + 8*(l>>4), n = l&15).
#pragma unroll
  for (int g = 0; g < 4; ++g) {
#pragma unroll
    for (int r = 0; r < 8; ++r) {
      int m = r + ((lane >> 4) << 3);
      part[wv][g][m * 16 + r15] = acc[g][r];
    }
  }
  __syncthreads();

  // Reduce 4 partials + bias, then fused LSTM cell update.
  for (int e = threadIdx.x; e < 256; e += 128) {
    int m = e >> 4;
    int n = e & 15;
    float gv[4];
#pragma unroll
    for (int g = 0; g < 4; ++g) {
      gv[g] = bias[g * NHID + n0 + n]
            + part[0][g][e] + part[1][g][e] + part[2][g][e] + part[3][g][e];
    }
    size_t idx = (size_t)m * NHID + n0 + n;
    float iv = sigmoidf_(gv[0]);
    float fv = sigmoidf_(gv[1]);
    float gg = tanhf(gv[2]);
    float ov = sigmoidf_(gv[3]);
    float cn = fv * C[idx] + iv * gg;
    float hn = ov * tanhf(cn);
    C[idx]  = cn;
    Hf[idx] = hn;
    _Float16 hh = (_Float16)hn;
    Hnext[idx] = hh;
    ys_h[idx]  = hh;
    if (ys_f) ys_f[idx] = hn;
  }
}

// ---------------------------------------------------------------------------
// Decoder GEMM: decoded = out @ Wdec^T + bdec   (1024 x 10000, K = 1024)
// Block = 4 waves, all sharing one M tile (mt).  The 16x32 A tile for each
// k-block is staged into LDS with CDNA5 async global->LDS copies (double
// buffered, ASYNCcnt-pipelined).  Each wave computes a 16x80 strip:
// 5 WMMAs per k-block against one shared A tile.
// Grid: 64 mt x 32 n-group-blocks; ngroup = 0..124 (125 groups of 80 cols).
// ---------------------------------------------------------------------------
__global__ void decoder_gemm_kernel(const _Float16* __restrict__ A,   // (1024,1024) f16
                                    const _Float16* __restrict__ W,   // (NTOK,1024) f16
                                    const float*    __restrict__ bias,// (NTOK)
                                    float*          __restrict__ D) { // (1024,NTOK)
  __shared__ _Float16 abuf[2][16 * 32];

  const int lane = threadIdx.x & 31;
  const int wv   = threadIdx.x >> 5;
  const int mt   = blockIdx.x >> 5;        // 0..63
  const int ngg  = blockIdx.x & 31;        // 0..31
  const int ngroup = ngg * 4 + wv;         // 0..127
  const bool active = (ngroup < (NTOK / 80));   // 125 groups
  const int r15 = lane & 15;
  const int nb  = ngroup * 80;

  const _Float16* abase = A + (size_t)(mt * 16) * NHID;

  v8f acc[5];
#pragma unroll
  for (int j = 0; j < 5; ++j) {
    float bi = active ? bias[nb + j * 16 + r15] : 0.0f;
    acc[j] = (v8f){bi, bi, bi, bi, bi, bi, bi, bi};
  }

  // prologue: stage A tile for kb = 0
  if (wv == 0) async_copy_a_tile(abase, NHID, 0, &abuf[0][0], lane);

  for (int i = 0; i < NHID / 32; ++i) {
    const int kb  = i * 32;
    const int cur = i & 1;
    if (wv == 0) {
      if (i + 1 < NHID / 32) {
        async_copy_a_tile(abase, NHID, kb + 32, &abuf[1 - cur][0], lane);
        wait_async_le2();        // current buffer's 2 copies are done
      } else {
        wait_async_0();
      }
    }
    __syncthreads();             // A tile [cur] visible to all 4 waves

    if (active) {
      v16h a = ld_tile_lds(&abuf[cur][0], lane);
#pragma unroll
      for (int j = 0; j < 5; ++j) {
        const _Float16* brow = W + (size_t)(nb + j * 16 + r15) * NHID;
        __builtin_prefetch(brow + kb + 256, 0, 1);   // global_prefetch_b8
        v16h b = ld_tile(brow, kb, lane);
        acc[j] = __builtin_amdgcn_wmma_f32_16x16x32_f16(
            false, a, false, b, (short)0, acc[j], false, false);
      }
    }
    __syncthreads();             // reads done before buffer is overwritten
  }

  if (active) {
#pragma unroll
    for (int j = 0; j < 5; ++j) {
#pragma unroll
      for (int r = 0; r < 8; ++r) {
        int m = mt * 16 + r + ((lane >> 4) << 3);
        D[(size_t)m * NTOK + nb + j * 16 + r15] = acc[j][r];
      }
    }
  }
}

// ---------------------------------------------------------------------------
// class_dist = softmax(out @ Wcl^T + bcl); one wave per row, lane-strided K,
// shfl_xor wave reduction.
// ---------------------------------------------------------------------------
__global__ void classdist_kernel(const float* __restrict__ out,  // (1024,NHID)
                                 const float* __restrict__ Wcl,  // (NCLASS,NHID)
                                 const float* __restrict__ bcl,
                                 float* __restrict__ cd) {       // (1024,NCLASS)
  const int lane = threadIdx.x & 31;
  const int row  = blockIdx.x * (blockDim.x >> 5) + (threadIdx.x >> 5);
  if (row >= ROWS) return;
  const float* x = out + (size_t)row * NHID;
  float a[NCLASS] = {0.f, 0.f, 0.f, 0.f};
  for (int k = lane; k < NHID; k += 32) {
    float xv = x[k];
#pragma unroll
    for (int j = 0; j < NCLASS; ++j) a[j] += xv * Wcl[j * NHID + k];
  }
#pragma unroll
  for (int j = 0; j < NCLASS; ++j) {
    for (int m = 16; m > 0; m >>= 1) a[j] += __shfl_xor(a[j], m, 32);
  }
  if (lane == 0) {
#pragma unroll
    for (int j = 0; j < NCLASS; ++j) a[j] += bcl[j];
    float mx = fmaxf(fmaxf(a[0], a[1]), fmaxf(a[2], a[3]));
    float s = 0.f;
#pragma unroll
    for (int j = 0; j < NCLASS; ++j) { a[j] = expf(a[j] - mx); s += a[j]; }
    float inv = 1.0f / s;
#pragma unroll
    for (int j = 0; j < NCLASS; ++j) cd[(size_t)row * NCLASS + j] = a[j] * inv;
  }
}

// ---------------------------------------------------------------------------
// Masked mixture-of-softmaxes.  One block per row; 3 passes over the 10000
// logits: masked max, masked sum-exp, weighted probability write.
// exp(-1e9 - max) underflows to 0, matching the reference's where(-1e9).
// ---------------------------------------------------------------------------
__global__ void mixture_kernel(const float* __restrict__ D,          // (1024,NTOK)
                               const unsigned char* __restrict__ mask,// (NCLASS,NTOK)
                               const float* __restrict__ cd,         // (1024,NCLASS)
                               float* __restrict__ outp) {           // (1024,NTOK)
  const int b   = blockIdx.x;
  const int tid = threadIdx.x;          // 256 threads
  __shared__ float sred[NCLASS][256];
  __shared__ float smax[NCLASS];
  __shared__ float sinv[NCLASS];

  const float* drow = D + (size_t)b * NTOK;

  float mx[NCLASS];
#pragma unroll
  for (int j = 0; j < NCLASS; ++j) mx[j] = -3.4e38f;
  for (int k = tid; k < NTOK; k += 256) {
    float d = drow[k];
#pragma unroll
    for (int j = 0; j < NCLASS; ++j) {
      float cand = mask[j * NTOK + k] ? -1e9f : d;
      mx[j] = fmaxf(mx[j], cand);
    }
  }
#pragma unroll
  for (int j = 0; j < NCLASS; ++j) sred[j][tid] = mx[j];
  __syncthreads();
  for (int s2 = 128; s2 > 0; s2 >>= 1) {
    if (tid < s2) {
#pragma unroll
      for (int j = 0; j < NCLASS; ++j)
        sred[j][tid] = fmaxf(sred[j][tid], sred[j][tid + s2]);
    }
    __syncthreads();
  }
  if (tid < NCLASS) smax[tid] = sred[tid][0];
  __syncthreads();

  float sm[NCLASS] = {0.f, 0.f, 0.f, 0.f};
  for (int k = tid; k < NTOK; k += 256) {
    float d = drow[k];
#pragma unroll
    for (int j = 0; j < NCLASS; ++j) {
      float cand = mask[j * NTOK + k] ? -1e9f : d;
      sm[j] += expf(cand - smax[j]);
    }
  }
#pragma unroll
  for (int j = 0; j < NCLASS; ++j) sred[j][tid] = sm[j];
  __syncthreads();
  for (int s2 = 128; s2 > 0; s2 >>= 1) {
    if (tid < s2) {
#pragma unroll
      for (int j = 0; j < NCLASS; ++j)
        sred[j][tid] += sred[j][tid + s2];
    }
    __syncthreads();
  }
  if (tid < NCLASS) sinv[tid] = cd[(size_t)b * NCLASS + tid] / sred[tid][0];
  __syncthreads();

  for (int k = tid; k < NTOK; k += 256) {
    float d = drow[k];
    float p = 0.f;
#pragma unroll
    for (int j = 0; j < NCLASS; ++j) {
      if (!mask[j * NTOK + k]) p += sinv[j] * expf(d - smax[j]);
    }
    outp[(size_t)b * NTOK + k] = p;
  }
}

// ---------------------------------------------------------------------------
// Copy final h/c stacks ([2][B][NHID] each) into the output tail.
// ---------------------------------------------------------------------------
__global__ void state_out_kernel(const float* __restrict__ Hf,
                                 const float* __restrict__ Cf,
                                 float* __restrict__ out_h,
                                 float* __restrict__ out_c) {
  int i = blockIdx.x * blockDim.x + threadIdx.x;
  if (i >= 2 * BATCH * NHID) return;
  out_h[i] = Hf[i];
  out_c[i] = Cf[i];
}

// ---------------------------------------------------------------------------
// Host driver
// ---------------------------------------------------------------------------
extern "C" void kernel_launch(void* const* d_in, const int* in_sizes, int n_in,
                              void* d_out, int out_size, void* d_ws, size_t ws_size,
                              hipStream_t stream) {
  (void)in_sizes; (void)n_in; (void)out_size; (void)ws_size;

  const int*   words = (const int*)  d_in[0];
  const unsigned char* cmask = (const unsigned char*)d_in[1];  // bool array
  const float* h0    = (const float*)d_in[2];
  const float* c0    = (const float*)d_in[3];
  const float* emb   = (const float*)d_in[4];
  const float* Wih0  = (const float*)d_in[5];
  const float* Whh0  = (const float*)d_in[6];
  const float* bih0  = (const float*)d_in[7];
  const float* bhh0  = (const float*)d_in[8];
  const float* Wih1  = (const float*)d_in[9];
  const float* Whh1  = (const float*)d_in[10];
  const float* bih1  = (const float*)d_in[11];
  const float* bhh1  = (const float*)d_in[12];
  const float* Wdec  = (const float*)d_in[13];
  const float* bdec  = (const float*)d_in[14];
  const float* Wcl   = (const float*)d_in[15];
  const float* bcl   = (const float*)d_in[16];

  // ---- workspace layout ----
  char* ws = (char*)d_ws;
  size_t off = 0;
  auto alloc = [&](size_t bytes) -> size_t {
    size_t o = off;
    off = (off + bytes + 255) & ~(size_t)255;
    return o;
  };
  size_t o_wih0h = alloc((size_t)NGATE * NINP * 2);
  size_t o_whh0h = alloc((size_t)NGATE * NHID * 2);
  size_t o_wih1h = alloc((size_t)NGATE * NHID * 2);
  size_t o_whh1h = alloc((size_t)NGATE * NHID * 2);
  size_t o_wdech = alloc((size_t)NTOK * NHID * 2);
  size_t o_xemb  = alloc((size_t)S_LEN * BATCH * NINP * 2);
  size_t o_ys0h  = alloc((size_t)S_LEN * BATCH * NHID * 2);
  size_t o_ys1h  = alloc((size_t)S_LEN * BATCH * NHID * 2);
  size_t o_ys1f  = alloc((size_t)S_LEN * BATCH * NHID * 4);
  size_t o_hf    = alloc((size_t)2 * BATCH * NHID * 4);
  size_t o_cf    = alloc((size_t)2 * BATCH * NHID * 4);
  size_t o_hh    = alloc((size_t)2 * 2 * BATCH * NHID * 2);  // [layer][ping][B*NHID]
  size_t o_b0    = alloc((size_t)NGATE * 4);
  size_t o_b1    = alloc((size_t)NGATE * 4);
  size_t o_dec   = alloc((size_t)ROWS * NTOK * 4);

  _Float16* wih0h = (_Float16*)(ws + o_wih0h);
  _Float16* whh0h = (_Float16*)(ws + o_whh0h);
  _Float16* wih1h = (_Float16*)(ws + o_wih1h);
  _Float16* whh1h = (_Float16*)(ws + o_whh1h);
  _Float16* wdech = (_Float16*)(ws + o_wdech);
  _Float16* xemb  = (_Float16*)(ws + o_xemb);
  _Float16* ys0h  = (_Float16*)(ws + o_ys0h);
  _Float16* ys1h  = (_Float16*)(ws + o_ys1h);
  float*    ys1f  = (float*)   (ws + o_ys1f);
  float*    Hf    = (float*)   (ws + o_hf);
  float*    Cf    = (float*)   (ws + o_cf);
  _Float16* Hh    = (_Float16*)(ws + o_hh);
  float*    b0    = (float*)   (ws + o_b0);
  float*    b1    = (float*)   (ws + o_b1);
  float*    dec   = (float*)   (ws + o_dec);

  float* out_sp = (float*)d_out;                         // (1024, NTOK)
  float* out_cd = out_sp + (size_t)ROWS * NTOK;          // (1024, NCLASS)
  float* out_h  = out_cd + (size_t)ROWS * NCLASS;        // (2, B, NHID)
  float* out_c  = out_h  + (size_t)2 * BATCH * NHID;     // (2, B, NHID)

  // ---- weight down-conversion (f32 -> f16) ----
  auto cvt = [&](const float* s, _Float16* d, int n) {
    f2h_kernel<<<(n + 255) / 256, 256, 0, stream>>>(s, d, n);
  };
  cvt(Wih0, wih0h, NGATE * NINP);
  cvt(Whh0, whh0h, NGATE * NHID);
  cvt(Wih1, wih1h, NGATE * NHID);
  cvt(Whh1, whh1h, NGATE * NHID);
  cvt(Wdec, wdech, NTOK * NHID);

  bias_sum_kernel<<<(NGATE + 255) / 256, 256, 0, stream>>>(bih0, bhh0, b0, NGATE);
  bias_sum_kernel<<<(NGATE + 255) / 256, 256, 0, stream>>>(bih1, bhh1, b1, NGATE);

  // ---- embedding gather ----
  {
    int n = S_LEN * BATCH * NINP;
    embed_kernel<<<(n + 255) / 256, 256, 0, stream>>>(words, emb, xemb);
  }

  // ---- LSTM state init ----
  init_state_kernel<<<(2 * BATCH * NHID + 255) / 256, 256, 0, stream>>>(
      h0, c0, Hf, Cf, Hh);

  const size_t STEP = (size_t)BATCH * NHID;  // 16384 elements per step slice
  const dim3 lstm_grid(64), lstm_block(128); // 64 tiles x (4-way split-K)

  // ---- layer 0: K = NINP, outputs f16 sequence ys0h ----
  for (int t = 0; t < S_LEN; ++t) {
    _Float16* hprev = Hh + 0 * 2 * STEP + (size_t)(t & 1) * STEP;
    _Float16* hnext = Hh + 0 * 2 * STEP + (size_t)((t + 1) & 1) * STEP;
    lstm_step_kernel<<<lstm_grid, lstm_block, 0, stream>>>(
        xemb + (size_t)t * BATCH * NINP, NINP, wih0h, whh0h, hprev, b0,
        Cf, Hf, hnext, ys0h + (size_t)t * STEP, (float*)nullptr);
  }

  // ---- layer 1: K = NHID, outputs f16 + f32 sequences ----
  for (int t = 0; t < S_LEN; ++t) {
    _Float16* hprev = Hh + 1 * 2 * STEP + (size_t)(t & 1) * STEP;
    _Float16* hnext = Hh + 1 * 2 * STEP + (size_t)((t + 1) & 1) * STEP;
    lstm_step_kernel<<<lstm_grid, lstm_block, 0, stream>>>(
        ys0h + (size_t)t * STEP, NHID, wih1h, whh1h, hprev, b1,
        Cf + STEP, Hf + STEP, hnext,
        ys1h + (size_t)t * STEP, ys1f + (size_t)t * STEP);
  }

  // ---- decoder GEMM: 64 mt x 32 ngroup-blocks, async-LDS staged A ----
  decoder_gemm_kernel<<<64 * 32, 128, 0, stream>>>(ys1h, wdech, bdec, dec);

  // ---- class distribution ----
  classdist_kernel<<<ROWS / 4, 128, 0, stream>>>(ys1f, Wcl, bcl, out_cd);

  // ---- masked mixture of softmaxes ----
  mixture_kernel<<<ROWS, 256, 0, stream>>>(dec, cmask, out_cd, out_sp);

  // ---- final h/c stacks ----
  state_out_kernel<<<(2 * BATCH * NHID + 255) / 256, 256, 0, stream>>>(
      Hf, Cf, out_h, out_c);
}
